// CudaTriangleAttention_36575941493402
// MI455X (gfx1250) — matrix operand order repaired and checked by reading, not compiled, not run on hardware
//
#include <hip/hip_runtime.h>
#include <hip/hip_bf16.h>

// Fused triangle (starting-node) attention for MI455X / gfx1250 (wave32, WMMA).
// One workgroup per row i; all intermediates live in 208KB of the 320KB WGP LDS.
// Weights are pre-swizzled (bf16, per-lane B-operand layout) into d_ws by a
// setup kernel so every WMMA B tile is two contiguous global_load_b128 per lane.

#define C_INX   128
#define HDX     128
#define NJ      256
#define NI      256
#define NHEADS  4
#define DHEAD   32
#define NWAVES  16
#define BLOCK   (NWAVES * 32)
#define W_ELEMS (C_INX * HDX)   // 16384 bf16 per swizzled weight

typedef unsigned short ushort_t;
typedef __bf16 bf16_t;
typedef __attribute__((ext_vector_type(16))) bf16_t v16bf;
typedef __attribute__((ext_vector_type(8)))  float  v8f;

// Native f32 -> bf16 (gfx1250 has hardware bf16 converts; clang lowers the
// fptrunc to v_cvt_*bf16_f32 instead of a 3-op integer RNE emulation).
static __device__ __forceinline__ ushort_t f2bf(float f) {
  bf16_t b = (bf16_t)f;
  return __builtin_bit_cast(ushort_t, b);
}

union AB32 { uint4 q[2]; v16bf v; ushort_t h[16]; };

// A operand (16x32 bf16), row-major source [.][ld]: lane holds row = lane%16,
// K = 16*(lane/16) + e  ->  two contiguous 16B chunks.
static __device__ __forceinline__ v16bf load_tileA(const ushort_t* base, int ld,
                                                   int r0, int k0, int lane) {
  const ushort_t* p = base + (size_t)(r0 + (lane & 15)) * ld + k0 + 16 * (lane >> 4);
  AB32 u;
  u.q[0] = *(const uint4*)p;
  u.q[1] = *(const uint4*)(p + 8);
  return u.v;
}

// B operand (32x16 bf16) from LDS: lane holds column n = lane%16, K = 16*(lane/16)+e
// -> 16 contiguous bf16 when the K axis is contiguous in memory.
static __device__ __forceinline__ v16bf load_tileB16(const ushort_t* p) {
  AB32 u;
  u.q[0] = ((const uint4*)p)[0];
  u.q[1] = ((const uint4*)p)[1];
  return u.v;
}

// B operand from a pre-swizzled bf16 weight in d_ws: 32 contiguous bytes per lane.
static __device__ __forceinline__ v16bf load_tileB_ws(const ushort_t* __restrict__ Wsw,
                                                      int kk, int n, int lane) {
  const ushort_t* p = Wsw + ((((kk * 8 + n) * 32) + lane) << 4);
  AB32 u;
  u.q[0] = ((const uint4*)p)[0];
  u.q[1] = ((const uint4*)p)[1];
  return u.v;
}

static __device__ __forceinline__ v8f wmma_bf16(v16bf a, v16bf b, v8f c) {
  return __builtin_amdgcn_wmma_f32_16x16x32_bf16(false, a, false, b, (short)0, c,
                                                 false, false);
}

static __device__ __forceinline__ float red_max16(float x) {
#pragma unroll
  for (int m = 1; m <= 8; m <<= 1) x = fmaxf(x, __shfl_xor(x, m, 32));
  return x;
}
static __device__ __forceinline__ float red_sum16(float x) {
#pragma unroll
  for (int m = 1; m <= 8; m <<= 1) x += __shfl_xor(x, m, 32);
  return x;
}

// ---- Setup: convert+swizzle the 5 weights into per-lane B-operand layout ----
// ws[widx][((kk*8+n)*32 + lane)*16 + e] = bf16(W[kk*32 + 16*(lane/16) + e][n*16 + lane%16])
__global__ void __launch_bounds__(256)
swizzle_weights_kernel(const float* __restrict__ Wq, const float* __restrict__ Wk,
                       const float* __restrict__ Wv, const float* __restrict__ Wg,
                       const float* __restrict__ Wo, ushort_t* __restrict__ ws) {
  const int idx  = blockIdx.x * 256 + threadIdx.x;   // 0 .. 16383
  const int widx = blockIdx.y;                       // 0 .. 4
  const int e    = idx & 15;
  const int lane = (idx >> 4) & 31;
  const int n    = (idx >> 9) & 7;
  const int kk   = idx >> 12;
  const float* W = (widx == 0) ? Wq : (widx == 1) ? Wk : (widx == 2) ? Wv
                  : (widx == 3) ? Wg : Wo;
  const int row = kk * 32 + 16 * (lane >> 4) + e;
  const int col = n * 16 + (lane & 15);
  ws[(size_t)widx * W_ELEMS + idx] = f2bf(W[(size_t)row * HDX + col]);
}

__global__ void __launch_bounds__(BLOCK)
tri_attn_kernel(const float* __restrict__ x, const unsigned char* __restrict__ mask,
                const float* __restrict__ ln_g, const float* __restrict__ ln_b,
                const ushort_t* __restrict__ wsw,   // 5 swizzled weights (q,k,v,g,o)
                const float* __restrict__ bg, const float* __restrict__ bo,
                float* __restrict__ out) {
  extern __shared__ ushort_t smem[];
  ushort_t* xn   = smem;                 // [256][128] bf16  (64KB) normed input
  ushort_t* kog  = xn + NJ * C_INX;      // [256][128] bf16  (64KB) K, reused as o*g
  ushort_t* vt   = kog + NJ * HDX;       // [128][256] bf16  (64KB) V transposed
  ushort_t* wscr = vt + HDX * NJ;        // 16 waves * 512   (16KB) C->A scratch

  const ushort_t* wsWq = wsw + 0 * W_ELEMS;
  const ushort_t* wsWk = wsw + 1 * W_ELEMS;
  const ushort_t* wsWv = wsw + 2 * W_ELEMS;
  const ushort_t* wsWg = wsw + 3 * W_ELEMS;
  const ushort_t* wsWo = wsw + 4 * W_ELEMS;

  const int i    = blockIdx.x;
  const int tid  = threadIdx.x;
  const int lane = tid & 31;
  const int w    = tid >> 5;
  const int r0   = w * 16;               // this wave's 16-row tile
  const int nn   = lane & 15;            // C-tile column
  const int mhl  = 8 * (lane >> 4);      // C-tile row offset for this lane half
  ushort_t* qs   = wscr + w * 512;       // wave-private [16][32] bf16 scratch

  v8f zero8 = {0.f, 0.f, 0.f, 0.f, 0.f, 0.f, 0.f, 0.f};

  // ---------------- Stage A: LayerNorm -> xn (bf16) ----------------
  {
    const float4 gv = ((const float4*)ln_g)[lane];
    const float4 bv = ((const float4*)ln_b)[lane];
#pragma unroll 1
    for (int p = 0; p < 16; ++p) {
      const int j = r0 + p;
      const float4 xv = ((const float4*)(x + ((size_t)i * NJ + j) * C_INX))[lane];
      float s  = xv.x + xv.y + xv.z + xv.w;
      float ss = xv.x * xv.x + xv.y * xv.y + xv.z * xv.z + xv.w * xv.w;
#pragma unroll
      for (int m = 1; m <= 16; m <<= 1) {
        s  += __shfl_xor(s, m, 32);
        ss += __shfl_xor(ss, m, 32);
      }
      const float mu  = s * (1.0f / C_INX);
      const float var = ss * (1.0f / C_INX) - mu * mu;
      const float rs  = rsqrtf(var + 1e-5f);
      const ushort_t h0 = f2bf((xv.x - mu) * rs * gv.x + bv.x);
      const ushort_t h1 = f2bf((xv.y - mu) * rs * gv.y + bv.y);
      const ushort_t h2 = f2bf((xv.z - mu) * rs * gv.z + bv.z);
      const ushort_t h3 = f2bf((xv.w - mu) * rs * gv.w + bv.w);
      uint2 pk;
      pk.x = (unsigned)h0 | ((unsigned)h1 << 16);
      pk.y = (unsigned)h2 | ((unsigned)h3 << 16);
      *(uint2*)(xn + (size_t)j * C_INX + lane * 4) = pk;
    }
  }
  __syncthreads();

  // ---------------- Stage B: K (row-major) and V^T projections ----------------
#pragma unroll 1
  for (int n = 0; n < 8; ++n) {
    v8f acck = zero8, accv = zero8;
#pragma unroll
    for (int kk = 0; kk < 4; ++kk) {
      const v16bf a = load_tileA(xn, C_INX, r0, kk * 32, lane);
      acck = wmma_bf16(a, load_tileB_ws(wsWk, kk, n, lane), acck);
      accv = wmma_bf16(a, load_tileB_ws(wsWv, kk, n, lane), accv);
    }
    const int col = n * 16 + nn;
#pragma unroll
    for (int r = 0; r < 8; ++r)
      kog[(size_t)(r0 + r + mhl) * HDX + col] = f2bf(acck[r]);
    AB32 pv;
#pragma unroll
    for (int r = 0; r < 8; ++r) pv.h[r] = f2bf(accv[r]);  // 8 consecutive keys
    *(uint4*)(vt + (size_t)col * NJ + r0 + mhl) = pv.q[0];
  }
  __syncthreads();

  // ---- mask bias, hoisted (keys handled by this lane are fixed per kb) ----
  float mb0[8], mb1[8];
#pragma unroll
  for (int kb = 0; kb < 8; ++kb) {
    mb0[kb] = mask[(size_t)i * NJ + kb * 32 + nn]      ? 0.0f : -1.0e9f;
    mb1[kb] = mask[(size_t)i * NJ + kb * 32 + 16 + nn] ? 0.0f : -1.0e9f;
  }

  // ---------------- Stage C: flash attention + gate, per head ----------------
  const float qscale = 0.17677669529663689f;  // 1/sqrt(32), folded into q
#pragma unroll 1
  for (int h = 0; h < NHEADS; ++h) {
    const int d0 = h * DHEAD;
    const int nq0 = d0 >> 4, nq1 = nq0 + 1;   // weight column-tile indices

    // q tile + gate tile on demand (C layout); shares the xn A tiles
    v8f q0 = zero8, q1 = zero8, g0 = zero8, g1 = zero8;
#pragma unroll
    for (int kk = 0; kk < 4; ++kk) {
      const v16bf a = load_tileA(xn, C_INX, r0, kk * 32, lane);
      q0 = wmma_bf16(a, load_tileB_ws(wsWq, kk, nq0, lane), q0);
      q1 = wmma_bf16(a, load_tileB_ws(wsWq, kk, nq1, lane), q1);
      g0 = wmma_bf16(a, load_tileB_ws(wsWg, kk, nq0, lane), g0);
      g1 = wmma_bf16(a, load_tileB_ws(wsWg, kk, nq1, lane), g1);
    }
#pragma unroll
    for (int r = 0; r < 8; ++r) {
      qs[(r + mhl) * 32 + nn]      = f2bf(q0[r] * qscale);
      qs[(r + mhl) * 32 + 16 + nn] = f2bf(q1[r] * qscale);
    }
    const v16bf qa = load_tileA(qs, 32, 0, 0, lane);

    v8f o0 = zero8, o1 = zero8;
    float mrun[8], lrun[8];
#pragma unroll
    for (int r = 0; r < 8; ++r) { mrun[r] = -1e30f; lrun[r] = 0.0f; }

#pragma unroll 1
    for (int kb = 0; kb < 8; ++kb) {
      const int key0 = kb * 32;
      const v16bf bk0 = load_tileB16(kog + (size_t)(key0 + nn) * HDX + d0 + 16 * (lane >> 4));
      const v16bf bk1 = load_tileB16(kog + (size_t)(key0 + 16 + nn) * HDX + d0 + 16 * (lane >> 4));
      v8f s0 = wmma_bf16(qa, bk0, zero8);
      v8f s1 = wmma_bf16(qa, bk1, zero8);

#pragma unroll
      for (int r = 0; r < 8; ++r) {
        const float v0 = s0[r] + mb0[kb];
        const float v1 = s1[r] + mb1[kb];
        const float mx   = red_max16(fmaxf(v0, v1));
        const float mnew = fmaxf(mrun[r], mx);
        const float alpha = __expf(mrun[r] - mnew);
        const float p0 = __expf(v0 - mnew);
        const float p1 = __expf(v1 - mnew);
        lrun[r] = lrun[r] * alpha + red_sum16(p0 + p1);
        mrun[r] = mnew;
        o0[r] *= alpha;
        o1[r] *= alpha;
        qs[(r + mhl) * 32 + nn]      = f2bf(p0);   // P -> A layout via scratch
        qs[(r + mhl) * 32 + 16 + nn] = f2bf(p1);
      }
      const v16bf pa  = load_tileA(qs, 32, 0, 0, lane);
      const v16bf bv0 = load_tileB16(vt + (size_t)(d0 + nn) * NJ + key0 + 16 * (lane >> 4));
      const v16bf bv1 = load_tileB16(vt + (size_t)(d0 + 16 + nn) * NJ + key0 + 16 * (lane >> 4));
      o0 = wmma_bf16(pa, bv0, o0);
      o1 = wmma_bf16(pa, bv1, o1);
    }

    // finalize: o/l * sigmoid(gate)
    const float bg0 = bg[d0 + nn], bg1 = bg[d0 + 16 + nn];
    float og0[8], og1[8];
#pragma unroll
    for (int r = 0; r < 8; ++r) {
      const float inv = 1.0f / lrun[r];
      const float gg0 = 1.0f / (1.0f + __expf(-(g0[r] + bg0)));
      const float gg1 = 1.0f / (1.0f + __expf(-(g1[r] + bg1)));
      og0[r] = o0[r] * inv * gg0;
      og1[r] = o1[r] * inv * gg1;
    }
    __syncthreads();  // everyone done reading K columns of this head
#pragma unroll
    for (int r = 0; r < 8; ++r) {  // overwrite dead K region with o*g (bf16)
      kog[(size_t)(r0 + r + mhl) * HDX + d0 + nn]      = f2bf(og0[r]);
      kog[(size_t)(r0 + r + mhl) * HDX + d0 + 16 + nn] = f2bf(og1[r]);
    }
  }
  __syncthreads();

  // ---------------- Stage D: out = x + (o*g) @ Wo + bo ----------------
#pragma unroll 1
  for (int n = 0; n < 8; ++n) {
    v8f acc = zero8;
#pragma unroll
    for (int kk = 0; kk < 4; ++kk)
      acc = wmma_bf16(load_tileA(kog, HDX, r0, kk * 32, lane),
                      load_tileB_ws(wsWo, kk, n, lane), acc);
    const int col = n * 16 + nn;
    const float bov = bo[col];
#pragma unroll
    for (int r = 0; r < 8; ++r) {
      const size_t idx = ((size_t)i * NJ + (r0 + r + mhl)) * C_INX + col;
      out[idx] = x[idx] + acc[r] + bov;
    }
  }
}

extern "C" void kernel_launch(void* const* d_in, const int* in_sizes, int n_in,
                              void* d_out, int out_size, void* d_ws, size_t ws_size,
                              hipStream_t stream) {
  (void)in_sizes; (void)n_in; (void)out_size; (void)ws_size;
  const float*         x    = (const float*)d_in[0];
  const unsigned char* mask = (const unsigned char*)d_in[1];
  const float*         ln_g = (const float*)d_in[2];
  const float*         ln_b = (const float*)d_in[3];
  const float*         Wq   = (const float*)d_in[4];
  const float*         Wk   = (const float*)d_in[5];
  const float*         Wv   = (const float*)d_in[6];
  const float*         Wg   = (const float*)d_in[7];
  const float*         bg   = (const float*)d_in[8];
  const float*         Wo   = (const float*)d_in[9];
  const float*         bo   = (const float*)d_in[10];
  float*    out = (float*)d_out;
  ushort_t* ws  = (ushort_t*)d_ws;   // 5 * 32KB swizzled bf16 weights

  swizzle_weights_kernel<<<dim3(W_ELEMS / 256, 5), dim3(256), 0, stream>>>(
      Wq, Wk, Wv, Wg, Wo, ws);

  const size_t smem_bytes =
      (size_t)(NJ * C_INX + NJ * HDX + HDX * NJ + NWAVES * 512) * sizeof(unsigned short);
  tri_attn_kernel<<<dim3(NI), dim3(BLOCK), smem_bytes, stream>>>(
      x, mask, ln_g, ln_b, ws, bg, bo, out);
}
